// Decoder_7017976561685
// MI455X (gfx1250) — matrix operand order repaired and checked by reading, not compiled, not run on hardware
//
#include <hip/hip_runtime.h>
#include <hip/hip_bf16.h>

#define N_TOK 8192
#define D_IN  32768
#define D_OUT 768

#define BM 128          // token rows per block
#define BN 128          // output cols per block
#define KS 32           // k per step (bf16 WMMA K)
#define LDSTRIDE_DW 20  // 40 bf16 per LDS row (32 + 8 pad) in dwords

typedef __attribute__((ext_vector_type(16))) __bf16   v16bf;
typedef __attribute__((ext_vector_type(2)))  __bf16   v2bf;
typedef __attribute__((ext_vector_type(8)))  float    v8f;
typedef __attribute__((ext_vector_type(4)))  float    f4;
typedef __attribute__((ext_vector_type(4)))  unsigned u4;

// Native fptrunc f32->bf16 (RNE). gfx1250 has hardware bf16 converts; let the
// backend pick v_cvt_pk_bf16_f32 / v_cvt_bf16 instead of a manual 6-op RNE.
static __device__ __forceinline__ unsigned pack_bf16(float a, float b) {
  v2bf r;
  r.x = (__bf16)a;
  r.y = (__bf16)b;
  return __builtin_bit_cast(unsigned, r);
}

static __device__ __forceinline__ v16bf load_frag(const unsigned* lo, const unsigned* hi) {
  union U { u4 u[2]; v16bf v; } r;
  r.u[0] = *(const u4*)lo;
  r.u[1] = *(const u4*)hi;
  return r.v;
}

__global__ __launch_bounds__(256, 1)
void sae_decode_wmma_bf16(const float* __restrict__ x,
                          const float* __restrict__ W,
                          const float* __restrict__ bpre,
                          float* __restrict__ out) {
  // double-buffered bf16 tiles, padded rows (20 dwords = 40 bf16)
  __shared__ unsigned Alds[2][BM * LDSTRIDE_DW];
  __shared__ unsigned Blds[2][BN * LDSTRIDE_DW];

  const int tid  = threadIdx.x;
  const int lane = tid & 31;
  const int wave = tid >> 5;
  const int wm   = wave >> 2;        // 0..1 : wave row  (64 tokens each)
  const int wn   = wave & 3;         // 0..3 : wave col  (32 outputs each)

  const int row0 = blockIdx.y * BM;  // token block
  const int col0 = blockIdx.x * BN;  // output-channel block (rows of W)

  // ---- loader mapping: 8 threads cover one 32-wide k-slice of a row ----
  const int chunk = tid & 7;         // which float4 of the 32 k's
  const int r0    = tid >> 3;        // 0..31 ; rows r0 + {0,32,64,96}

  const float* xg = x + (size_t)row0 * D_IN + chunk * 4;
  const float* wg = W + (size_t)col0 * D_IN + chunk * 4;

  v8f acc[4][2] = {};                // 8 WMMA accumulators per wave (fp32)

  // fragment LDS addressing (ISA §7.12.2 layouts)
  const int am       = lane & 15;
  const int aklo_dw  = (lane < 16) ? 0 : 4;   // A: K {0..7} vs {8..15} (then +16)
  const int bkr_dw   = (lane < 16) ? 0 : 4;   // B: K {0..7} vs {8..15} (then +16)

  f4 areg[4], breg[4], bp;

  // ---------------- prologue: fill buffer 0 ----------------
  bp = *(const f4*)(bpre + chunk * 4);
#pragma unroll
  for (int i = 0; i < 4; ++i) {
    areg[i] = *(const f4*)(xg + (size_t)(r0 + 32 * i) * D_IN);
    breg[i] = *(const f4*)(wg + (size_t)(r0 + 32 * i) * D_IN);
  }
#pragma unroll
  for (int i = 0; i < 4; ++i) {
    unsigned* da = &Alds[0][(r0 + 32 * i) * LDSTRIDE_DW + chunk * 2];
    da[0] = pack_bf16(areg[i].x - bp.x, areg[i].y - bp.y);
    da[1] = pack_bf16(areg[i].z - bp.z, areg[i].w - bp.w);
    unsigned* db = &Blds[0][(r0 + 32 * i) * LDSTRIDE_DW + chunk * 2];
    db[0] = pack_bf16(breg[i].x, breg[i].y);
    db[1] = pack_bf16(breg[i].z, breg[i].w);
  }
  __syncthreads();

  const int NK = D_IN / KS;          // 1024 k-steps
  for (int kt = 0; kt < NK; ++kt) {
    const int cur = kt & 1;
    const int nxt = cur ^ 1;
    const bool have_next = (kt + 1 < NK);
    const int knext = (kt + 1) * KS;

    // issue next tile's global loads early: latency hides under WMMAs
    if (have_next) {
      bp = *(const f4*)(bpre + knext + chunk * 4);
#pragma unroll
      for (int i = 0; i < 4; ++i) {
        areg[i] = *(const f4*)(xg + knext + (size_t)(r0 + 32 * i) * D_IN);
        breg[i] = *(const f4*)(wg + knext + (size_t)(r0 + 32 * i) * D_IN);
      }
    }

    // ---------------- compute on buffer `cur` ----------------
    v16bf bfrag[2], afrag[4];
#pragma unroll
    for (int nt = 0; nt < 2; ++nt) {
      const unsigned* bb =
          &Blds[cur][(wn * 32 + nt * 16 + am) * LDSTRIDE_DW + bkr_dw];
      bfrag[nt] = load_frag(bb, bb + 8);        // K lo-run + hi-run (+16 elems)
    }
#pragma unroll
    for (int mt = 0; mt < 4; ++mt) {
      const unsigned* ab =
          &Alds[cur][(wm * 64 + mt * 16 + am) * LDSTRIDE_DW + aklo_dw];
      afrag[mt] = load_frag(ab, ab + 8);        // K lo-run + hi-run (+16 elems)
    }
#pragma unroll
    for (int mt = 0; mt < 4; ++mt) {
#pragma unroll
      for (int nt = 0; nt < 2; ++nt) {
        acc[mt][nt] = __builtin_amdgcn_wmma_f32_16x16x32_bf16(
            false, afrag[mt], false, bfrag[nt], (short)0, acc[mt][nt],
            false, false);
      }
    }

    // convert + stage next tile into the other buffer (safe: all waves
    // finished reading it before the previous barrier)
    if (have_next) {
#pragma unroll
      for (int i = 0; i < 4; ++i) {
        unsigned* da = &Alds[nxt][(r0 + 32 * i) * LDSTRIDE_DW + chunk * 2];
        da[0] = pack_bf16(areg[i].x - bp.x, areg[i].y - bp.y);
        da[1] = pack_bf16(areg[i].z - bp.z, areg[i].w - bp.w);
        unsigned* db = &Blds[nxt][(r0 + 32 * i) * LDSTRIDE_DW + chunk * 2];
        db[0] = pack_bf16(breg[i].x, breg[i].y);
        db[1] = pack_bf16(breg[i].z, breg[i].w);
      }
    }
    __syncthreads();
  }

  // ---------------- epilogue: C/D layout per ISA ----------------
  // VGPR v holds element (M = v + 8*(lane>=16), N = lane&15)
  const int m_base = row0 + wm * 64 + ((lane >> 4) << 3);
  const int n_base = col0 + wn * 32 + (lane & 15);
#pragma unroll
  for (int mt = 0; mt < 4; ++mt) {
#pragma unroll
    for (int nt = 0; nt < 2; ++nt) {
      const int col = n_base + nt * 16;
#pragma unroll
      for (int v = 0; v < 8; ++v) {
        const int m = m_base + mt * 16 + v;
        out[(size_t)m * D_OUT + col] = acc[mt][nt][v];
      }
    }
  }
}

extern "C" void kernel_launch(void* const* d_in, const int* in_sizes, int n_in,
                              void* d_out, int out_size, void* d_ws, size_t ws_size,
                              hipStream_t stream) {
  (void)in_sizes; (void)n_in; (void)out_size; (void)d_ws; (void)ws_size;
  const float* x    = (const float*)d_in[0];
  const float* W    = (const float*)d_in[1];
  const float* bpre = (const float*)d_in[2];
  float* out        = (float*)d_out;

  dim3 grid(D_OUT / BN, N_TOK / BM);  // (6, 64)
  dim3 block(256);
  sae_decode_wmma_bf16<<<grid, block, 0, stream>>>(x, W, bpre, out);
}
